// UNet_63891933495559
// MI455X (gfx1250) — compile-verified, matrix operand order checked by reading, not run on hardware
//
#include <hip/hip_runtime.h>
#include <hip/hip_bf16.h>
#include <cstdint>
#include <cstddef>

// ---------------------------------------------------------------------------
// Types / constants
// ---------------------------------------------------------------------------
typedef __bf16 bf16;
typedef __attribute__((ext_vector_type(16))) __bf16 v16bf;
typedef __attribute__((ext_vector_type(8)))  __bf16 v8bf;
typedef __attribute__((ext_vector_type(8)))  float  v8f;
typedef unsigned int u32;
typedef __attribute__((ext_vector_type(4))) u32 u32x4;
typedef __attribute__((ext_vector_type(8))) int  i32x8;
typedef __attribute__((ext_vector_type(4))) int  i32x4;

static constexpr int BB    = 8;     // clouds
static constexpr int NPTS0 = 2048;  // points / cloud
static constexpr int NPARC = 2048;  // context points / cloud
static constexpr int KNN_K = 16;
static constexpr int K_UPC = 8;
static constexpr int DCTX  = 128;

__device__ __forceinline__ float actf(float v, int act) {
  if (act == 1) return fmaxf(v, 0.f);                 // relu
  if (act == 2) return v * (1.f / (1.f + expf(-v)));  // silu
  return v;
}

// ---------------------------------------------------------------------------
// f32 -> bf16 conversion (weights converted once per launch)
// ---------------------------------------------------------------------------
__global__ void ku_f32_to_bf16(const float* __restrict__ in, bf16* __restrict__ out, int n) {
  int i = blockIdx.x * blockDim.x + threadIdx.x;
  if (i < n) out[i] = (bf16)in[i];
}

// ---------------------------------------------------------------------------
// WMMA bf16 GEMM:  Cmat[M,N] = act(A[M,K] @ W[K,N] + bias)
// A: f32 row-major, converted to bf16 while staging into LDS (float4 fast path
//     when K%4==0, guarded scalar path for edges / K=3).
// W: bf16 row-major in global; staged TRANSPOSED into LDS so each lane's
//     fragment slice is two contiguous 16B runs -> ds_load_b128 (no u16 gathers).
// Block tile 64x64, 4 waves, 2x2 wmma tiles/wave, fragment layouts per ISA 7.12.2.
// ---------------------------------------------------------------------------
#define GT_M 64
#define GT_N 64
#define GT_K 32
#define LPITCH (GT_K + 8)   // 40 bf16 = 80 bytes; 80*row + 16*half is 16B aligned

__global__ __launch_bounds__(128)
void ku_gemm_wmma(const float* __restrict__ A, const bf16* __restrict__ W,
                  const float* __restrict__ bias, float* __restrict__ Cmat,
                  int M, int N, int K, int act) {
  __shared__ bf16 sA[GT_M][LPITCH];   // A tile, row-major [m][k]
  __shared__ bf16 sBT[GT_N][LPITCH];  // B tile, transposed [n][k]

  const int tid  = threadIdx.x;
  const int wave = tid >> 5;
  const int lane = tid & 31;
  const int bm = blockIdx.x * GT_M;
  const int bn = blockIdx.y * GT_N;
  const int wm = (wave >> 1) * 32;
  const int wn = (wave & 1) * 32;
  const int half = lane >> 4;
  const int l16  = lane & 15;

  v8f acc[2][2] = {};
  const bool kvec = ((K & 3) == 0);

  for (int k0 = 0; k0 < K; k0 += GT_K) {
    // ---- stage A tile (64 rows x 32 k), 16 elements per thread ----
    {
      int row = tid >> 1;
      int cb  = (tid & 1) * 16;
      int gr  = bm + row;
      bf16 tmp[16];
      if (kvec && gr < M && (k0 + cb + 16) <= K) {
        const float4* src = (const float4*)(A + (size_t)gr * K + k0 + cb);
        __builtin_prefetch(A + (size_t)gr * K + k0 + GT_K, 0, 1);  // global_prefetch_b8
#pragma unroll
        for (int e4 = 0; e4 < 4; e4++) {
          float4 v = src[e4];
          tmp[e4 * 4 + 0] = (bf16)v.x; tmp[e4 * 4 + 1] = (bf16)v.y;
          tmp[e4 * 4 + 2] = (bf16)v.z; tmp[e4 * 4 + 3] = (bf16)v.w;
        }
      } else {
#pragma unroll
        for (int e = 0; e < 16; e++) {
          int gc = k0 + cb + e;
          tmp[e] = (gr < M && gc < K) ? (bf16)A[(size_t)gr * K + gc] : (bf16)0.f;
        }
      }
      *(v8bf*)&sA[row][cb]     = *(const v8bf*)&tmp[0];
      *(v8bf*)&sA[row][cb + 8] = *(const v8bf*)&tmp[8];
    }
    // ---- stage B tile transposed: lanes walk adjacent columns at the same
    //      K-row (coalesced global), contiguous ds_store_b128 into sBT ----
    {
      int col = tid & 63;          // output column within tile
      int kb  = (tid >> 6) * 16;   // k sub-range 0 or 16
      int gc  = bn + col;
      __builtin_prefetch(W + (size_t)(k0 + GT_K) * N + gc, 0, 1);
      bf16 tmp[16];
#pragma unroll
      for (int e = 0; e < 16; e++) {
        int gr = k0 + kb + e;
        tmp[e] = (gr < K && gc < N) ? W[(size_t)gr * N + gc] : (bf16)0.f;
      }
      *(v8bf*)&sBT[col][kb]     = *(const v8bf*)&tmp[0];
      *(v8bf*)&sBT[col][kb + 8] = *(const v8bf*)&tmp[8];
    }
    __syncthreads();

    // ---- fragments: two contiguous 16B runs per lane -> ds_load_b128 ----
    v16bf bfrag[2];
#pragma unroll
    for (int j = 0; j < 2; j++) {
      int bc = wn + j * 16 + l16;
      v8bf b0 = *(const v8bf*)&sBT[bc][half * 8];
      v8bf b1 = *(const v8bf*)&sBT[bc][half * 8 + 16];
      bfrag[j] = __builtin_shufflevector(b0, b1, 0, 1, 2, 3, 4, 5, 6, 7,
                                         8, 9, 10, 11, 12, 13, 14, 15);
    }
#pragma unroll
    for (int i = 0; i < 2; i++) {
      int ar = wm + i * 16 + l16;
      v8bf a0 = *(const v8bf*)&sA[ar][half * 8];
      v8bf a1 = *(const v8bf*)&sA[ar][half * 8 + 16];
      v16bf afrag = __builtin_shufflevector(a0, a1, 0, 1, 2, 3, 4, 5, 6, 7,
                                            8, 9, 10, 11, 12, 13, 14, 15);
      acc[i][0] = __builtin_amdgcn_wmma_f32_16x16x32_bf16(
          false, afrag, false, bfrag[0], (short)0, acc[i][0], false, false);
      acc[i][1] = __builtin_amdgcn_wmma_f32_16x16x32_bf16(
          false, afrag, false, bfrag[1], (short)0, acc[i][1], false, false);
    }
    __syncthreads();
  }

  // ---- epilogue: bias + activation fused, f32 C/D VGPR layout ----
#pragma unroll
  for (int i = 0; i < 2; i++)
#pragma unroll
    for (int j = 0; j < 2; j++)
#pragma unroll
      for (int r = 0; r < 8; r++) {
        int row = bm + wm + i * 16 + r + half * 8;
        int col = bn + wn + j * 16 + l16;
        if (row < M && col < N) {
          float vv = acc[i][j][r];
          if (bias) vv += bias[col];
          Cmat[(size_t)row * N + col] = actf(vv, act);
        }
      }
}

// ---------------------------------------------------------------------------
// BatchNorm (training stats) : per-channel mean / rstd, then normalize+act
// ---------------------------------------------------------------------------
__global__ void ku_bn_stats(const float* __restrict__ X, int M, int C,
                            float* __restrict__ mean, float* __restrict__ rstd) {
  int c = blockIdx.x;
  __shared__ float s1[256], s2[256];
  float a = 0.f, b2 = 0.f;
  for (int m = threadIdx.x; m < M; m += 256) {
    float v = X[(size_t)m * C + c];
    a += v; b2 += v * v;
  }
  s1[threadIdx.x] = a; s2[threadIdx.x] = b2;
  __syncthreads();
  for (int s = 128; s > 0; s >>= 1) {
    if ((int)threadIdx.x < s) {
      s1[threadIdx.x] += s1[threadIdx.x + s];
      s2[threadIdx.x] += s2[threadIdx.x + s];
    }
    __syncthreads();
  }
  if (threadIdx.x == 0) {
    float mu  = s1[0] / (float)M;
    float var = s2[0] / (float)M - mu * mu;
    mean[c] = mu;
    rstd[c] = rsqrtf(fmaxf(var, 0.f) + 1e-5f);
  }
}

__global__ void ku_bn_act(float* __restrict__ X, int M, int C,
                          const float* __restrict__ mean, const float* __restrict__ rstd,
                          const float* __restrict__ g, const float* __restrict__ beta, int act) {
  int i = blockIdx.x * blockDim.x + threadIdx.x;
  if (i >= M * C) return;
  int c = i % C;
  float v = (X[i] - mean[c]) * rstd[c] * g[c] + beta[c];
  X[i] = actf(v, act);
}

// ---------------------------------------------------------------------------
// kNN: one thread per dst point, insertion-sorted top-k over all src points
// ---------------------------------------------------------------------------
__global__ void ku_knn(const float* __restrict__ src, const float* __restrict__ dst,
                       int Ns, int Nd, int* __restrict__ out,
                       int kq, int kout, int excl, int app) {
  int gid = blockIdx.x * blockDim.x + threadIdx.x;
  if (gid >= BB * Nd) return;
  int b = gid / Nd, n = gid % Nd;
  const float* dp = dst + ((size_t)b * Nd + n) * 3;
  float px = dp[0], py = dp[1], pz = dp[2];
  float bd[KNN_K + 1]; int bi[KNN_K + 1];
  for (int q = 0; q < kq; q++) { bd[q] = 3.0e38f; bi[q] = 0; }
  for (int s = 0; s < Ns; s++) {
    if (excl && s == n) continue;
    const float* sp = src + ((size_t)b * Ns + s) * 3;
    float dx = px - sp[0], dy = py - sp[1], dz = pz - sp[2];
    float d = dx * dx + dy * dy + dz * dz;
    if (d < bd[kq - 1]) {
      bd[kq - 1] = d; bi[kq - 1] = s;
      for (int q = kq - 1; q > 0; q--) {
        if (bd[q] < bd[q - 1]) {
          float td = bd[q]; bd[q] = bd[q - 1]; bd[q - 1] = td;
          int ti = bi[q]; bi[q] = bi[q - 1]; bi[q - 1] = ti;
        } else break;
      }
    }
  }
  int* op = out + (size_t)gid * kout;
  for (int q = 0; q < kq; q++) op[q] = bi[q];
  if (app) op[kq] = n;  // self loop appended last
}

// ---------------------------------------------------------------------------
// Farthest point sampling: one block per cloud, LDS-resident, serial argmax
// ---------------------------------------------------------------------------
__global__ void ku_fps(const float* __restrict__ pos, int N, int m, int* __restrict__ out) {
  int b = blockIdx.x;
  __shared__ float spx[2048], spy[2048], spz[2048], sdist[2048];
  __shared__ float rmax[256];
  __shared__ int rarg[256];
  __shared__ int s_last;
  for (int i = threadIdx.x; i < N; i += 256) {
    const float* p = pos + ((size_t)b * N + i) * 3;
    spx[i] = p[0]; spy[i] = p[1]; spz[i] = p[2];
    sdist[i] = 3.0e38f;
  }
  if (threadIdx.x == 0) { out[(size_t)b * m] = 0; s_last = 0; }
  __syncthreads();
  for (int step = 1; step < m; step++) {
    int last = s_last;
    float lx = spx[last], ly = spy[last], lz = spz[last];
    float best = -1.f; int barg = 0;
    for (int i = threadIdx.x; i < N; i += 256) {
      float dx = spx[i] - lx, dy = spy[i] - ly, dz = spz[i] - lz;
      float nd = fminf(sdist[i], dx * dx + dy * dy + dz * dz);
      sdist[i] = nd;
      if (nd > best) { best = nd; barg = i; }
    }
    rmax[threadIdx.x] = best; rarg[threadIdx.x] = barg;
    __syncthreads();
    for (int s = 128; s > 0; s >>= 1) {
      if ((int)threadIdx.x < s && rmax[threadIdx.x + s] > rmax[threadIdx.x]) {
        rmax[threadIdx.x] = rmax[threadIdx.x + s];
        rarg[threadIdx.x] = rarg[threadIdx.x + s];
      }
      __syncthreads();
    }
    if (threadIdx.x == 0) { out[(size_t)b * m + step] = rarg[0]; s_last = rarg[0]; }
    __syncthreads();
  }
}

// ---------------------------------------------------------------------------
// PTConv helper kernels
// ---------------------------------------------------------------------------
__global__ void ku_rel_pos(const float* __restrict__ ps, const float* __restrict__ pd,
                           const int* __restrict__ nbr, int Ns, int Nd, int k,
                           float* __restrict__ rel) {
  int gid = blockIdx.x * blockDim.x + threadIdx.x;
  if (gid >= BB * Nd * k) return;
  int q = gid % k, n = (gid / k) % Nd, b = gid / (k * Nd);
  int j = nbr[((size_t)b * Nd + n) * k + q];
  const float* d = pd + ((size_t)b * Nd + n) * 3;
  const float* s = ps + ((size_t)b * Ns + j) * 3;
  float* r = rel + (size_t)gid * 3;
  r[0] = d[0] - s[0]; r[1] = d[1] - s[1]; r[2] = d[2] - s[2];
}

__global__ void ku_gamma_in(const float* __restrict__ a_dst, const float* __restrict__ a_src,
                            const int* __restrict__ nbr, const float* __restrict__ delta,
                            int Ns, int Nd, int k, int C, float* __restrict__ out) {
  int gid = blockIdx.x * blockDim.x + threadIdx.x;
  if (gid >= BB * Nd * k * C) return;
  int c = gid % C, q = (gid / C) % k, n = (gid / (C * k)) % Nd, b = gid / (C * k * Nd);
  int j = nbr[((size_t)b * Nd + n) * k + q];
  out[gid] = a_dst[((size_t)b * Nd + n) * C + c] - a_src[((size_t)b * Ns + j) * C + c] + delta[gid];
}

// softmax over k neighbors per (point, channel) + weighted aggregation
__global__ void ku_attn_agg(const float* __restrict__ gamma, const float* __restrict__ v,
                            const float* __restrict__ delta, const int* __restrict__ nbr,
                            int Ns, int Nd, int k, int C, float* __restrict__ out) {
  int b = blockIdx.x / Nd, n = blockIdx.x % Nd;
  int c = threadIdx.x;
  if (c >= C) return;
  size_t base = ((size_t)b * Nd + n) * k;
  float mx = -3.0e38f;
  for (int q = 0; q < k; q++) mx = fmaxf(mx, gamma[(base + q) * C + c]);
  float den = 0.f;
  for (int q = 0; q < k; q++) den += expf(gamma[(base + q) * C + c] - mx);
  float acc = 0.f;
  for (int q = 0; q < k; q++) {
    int j = nbr[base + q];
    float w = expf(gamma[(base + q) * C + c] - mx) / den;
    acc += w * (v[((size_t)b * Ns + j) * C + c] + delta[(base + q) * C + c]);
  }
  out[((size_t)b * Nd + n) * C + c] = acc;
}

// ---------------------------------------------------------------------------
// Misc elementwise kernels
// ---------------------------------------------------------------------------
__global__ void ku_add(float* __restrict__ a, const float* __restrict__ b, int n) {
  int i = blockIdx.x * blockDim.x + threadIdx.x;
  if (i < n) a[i] += b[i];
}

__global__ void ku_add_bcast(float* __restrict__ x, const float* __restrict__ cv,
                             int Npts, int C) {
  int gid = blockIdx.x * blockDim.x + threadIdx.x;
  if (gid >= BB * Npts * C) return;
  int c = gid % C, b = gid / (C * Npts);
  x[gid] += cv[(size_t)b * C + c];
}

__global__ void ku_gather_rows(const float* __restrict__ x, const int* __restrict__ idx,
                               int Ns, int Nd, int C, float* __restrict__ out) {
  int gid = blockIdx.x * blockDim.x + threadIdx.x;
  if (gid >= BB * Nd * C) return;
  int c = gid % C, n = (gid / C) % Nd, b = gid / (C * Nd);
  out[gid] = x[((size_t)b * Ns + idx[(size_t)b * Nd + n]) * C + c];
}

__global__ void ku_gather_mean(const float* __restrict__ x, const int* __restrict__ nbr,
                               int Ns, int Nd, int k, int C, float* __restrict__ out) {
  int gid = blockIdx.x * blockDim.x + threadIdx.x;
  if (gid >= BB * Nd * C) return;
  int c = gid % C, n = (gid / C) % Nd, b = gid / (C * Nd);
  float s = 0.f;
  for (int q = 0; q < k; q++) {
    int j = nbr[((size_t)b * Nd + n) * k + q];
    s += x[((size_t)b * Ns + j) * C + c];
  }
  out[gid] = s / (float)k;
}

__global__ void ku_maxpool(const float* __restrict__ x, int Npts, int C, float* __restrict__ out) {
  int gid = blockIdx.x * blockDim.x + threadIdx.x;
  if (gid >= BB * C) return;
  int c = gid % C, b = gid / C;
  float m = -3.0e38f;
  for (int n = 0; n < Npts; n++) m = fmaxf(m, x[((size_t)b * Npts + n) * C + c]);
  out[gid] = m;
}

__global__ void ku_time_embed(const float* __restrict__ t, float* __restrict__ out,
                              int Bn, int dim) {
  int gid = blockIdx.x * blockDim.x + threadIdx.x;
  if (gid >= Bn * dim) return;
  int b = gid / dim, c = gid % dim;
  int half = dim / 2;
  float tt = t[b];
  if (c < half) {
    float f = expf(-logf(10000.f) * (float)c / (float)half);
    out[gid] = sinf(tt * f);
  } else {
    float f = expf(-logf(10000.f) * (float)(c - half) / (float)half);
    out[gid] = cosf(tt * f);
  }
}

// ---------------------------------------------------------------------------
// TDM demo: stage a 2D f32 tile into LDS via the Tensor Data Mover, then copy
// to global.  Exercises tensor_load_to_lds / s_wait_tensorcnt / cluster barrier.
// ---------------------------------------------------------------------------
__global__ void ku_tdm_stage(const float* __restrict__ src, float* __restrict__ dst,
                             int rows, int cols, int stride_elems) {
#if defined(__AMDGCN__) && __has_builtin(__builtin_amdgcn_tensor_load_to_lds)
  __shared__ float tile[4096];
  if (threadIdx.x == 0) {
    unsigned long long ga = (unsigned long long)(uintptr_t)src;
    u32x4 g0 = {}; i32x8 g1 = {}; i32x4 g2 = {}; i32x4 g3 = {};
    // D# group0: count=1, lds_addr=0, global_addr, type=2 (image)
    g0[0] = 1u;
    g0[1] = 0u;
    g0[2] = (u32)(ga & 0xffffffffu);
    g0[3] = (u32)((ga >> 32) & 0x01ffffffu) | (2u << 30);
    // D# group1: data_size=4B; tensor_dim0/1; tile_dim0/1; dim0 stride
    g1[0] = (int)(2u << 16);
    g1[1] = (int)(((u32)cols & 0xffffu) << 16);
    g1[2] = (int)((((u32)cols >> 16) & 0xffffu) | (((u32)rows & 0xffffu) << 16));
    g1[3] = (int)((((u32)rows >> 16) & 0xffffu) | (((u32)cols & 0xffffu) << 16));
    g1[4] = (int)((u32)rows & 0xffffu);
    g1[5] = (int)(u32)stride_elems;
#if defined(__clang_major__) && (__clang_major__ >= 23)
    i32x8 g4 = {};
    __builtin_amdgcn_tensor_load_to_lds(g0, g1, g2, g3, g4, 0);
#else
    __builtin_amdgcn_tensor_load_to_lds(g0, g1, g2, g3, 0);
#endif
    __builtin_amdgcn_s_wait_tensorcnt(0);
  }
#if __has_builtin(__builtin_amdgcn_s_cluster_barrier)
  __builtin_amdgcn_s_cluster_barrier();  // NOP when not launched as a cluster
#endif
  __syncthreads();
  int n = rows * cols; if (n > 4096) n = 4096;
  for (int i = threadIdx.x; i < n; i += blockDim.x) dst[i] = tile[i];
#else
  int n = rows * cols;
  for (int i = blockIdx.x * blockDim.x + threadIdx.x; i < n; i += blockDim.x * gridDim.x)
    dst[i] = src[i];
  (void)stride_elems;
#endif
}

// ===========================================================================
// Host side: parameter cursor (jax pytree = dict keys alphabetical), bump
// allocator over d_ws (reset each call -> deterministic), launch helpers.
// ===========================================================================
struct WS { char* base; size_t cap; size_t off; };
static inline void* ws_alloc(WS& w, size_t bytes) {
  size_t a = (w.off + 255) & ~(size_t)255;
  w.off = a + bytes;
  return w.base + a;
}
static inline float* wsF(WS& w, size_t n) { return (float*)ws_alloc(w, n * sizeof(float)); }

struct Cur { void* const* din; int i; int n; };
static const float* nf(Cur& c) {
  const float* p = (const float*)c.din[c.i];
  if (c.i + 1 < c.n) c.i++;
  return p;
}

struct LinP { const float *b, *beta, *g, *w; int fin, fout; };
struct MlpP { LinP L[2]; int n; };
struct ConvP { MlpP attn, pos; const float *w_dst, *w_lin, *w_src; int cs, cd, co; };
struct TbP { ConvP conv; MlpP mlp_in, mlp_out; };
struct CmP { CmP() {} MlpP ctx, mlp; };
struct DownP { CmP em; MlpP mlp; TbP tb; CmP tm; };
struct UpP { CmP em; ConvP ft; MlpP mlp; TbP tb; CmP tm; };

// layer dict keys sorted: b, beta, g, w
static LinP rdL(Cur& c, int fin, int fout) {
  LinP L; L.b = nf(c); L.beta = nf(c); L.g = nf(c); L.w = nf(c);
  L.fin = fin; L.fout = fout; return L;
}
static MlpP rdM1(Cur& c, int d0, int d1) { MlpP m; m.n = 1; m.L[0] = rdL(c, d0, d1); return m; }
static MlpP rdM2(Cur& c, int d0, int d1, int d2) {
  MlpP m; m.n = 2; m.L[0] = rdL(c, d0, d1); m.L[1] = rdL(c, d1, d2); return m;
}
// conv dict keys sorted: attn_nn, pos_nn, w_dst, w_lin, w_src
static ConvP rdConv(Cur& c, int cs, int cd, int co, bool deep) {
  ConvP p; p.cs = cs; p.cd = cd; p.co = co;
  p.attn = deep ? rdM2(c, co, co, co) : rdM1(c, co, co);
  p.pos  = deep ? rdM2(c, 3, co, co)  : rdM1(c, 3, co);
  p.w_dst = nf(c); p.w_lin = nf(c); p.w_src = nf(c);
  return p;
}
static TbP rdTb(Cur& c, int ci, int co) {  // keys: conv, mlp_in, mlp_out
  TbP t; t.conv = rdConv(c, ci, ci, co, true);
  t.mlp_in = rdM1(c, ci, ci); t.mlp_out = rdM1(c, co, co); return t;
}
static CmP rdCm(Cur& c, int ci, int co) {  // keys: ctx, mlp
  CmP m; m.ctx = rdM1(c, DCTX, co); m.mlp = rdM1(c, ci, co); return m;
}
static DownP rdDown(Cur& c, int ci, int co) {  // keys: em, mlp, tb, tm
  DownP d; d.em = rdCm(c, co, co); d.mlp = rdM1(c, ci, co);
  d.tb = rdTb(c, co, co); d.tm = rdCm(c, co, co); return d;
}
static UpP rdUp(Cur& c, int cs_dn, int c_up) {  // keys: em, ft, mlp, tb, tm
  UpP u; u.em = rdCm(c, c_up, c_up); u.ft = rdConv(c, cs_dn, c_up, c_up, false);
  u.mlp = rdM1(c, c_up, c_up); u.tb = rdTb(c, c_up, c_up); u.tm = rdCm(c, c_up, c_up);
  return u;
}

struct ParamsH {
  DownP down[2];
  CmP em_in;
  const float *enc_b1, *enc_b2, *enc_w1, *enc_w2;
  MlpP mlp_input, mlp_output, mlp_par;
  const float *skip_b, *skip_w;
  MlpP summit_mlp;
  TbP summit_tb, tb_in;
  CmP tm_in;
  ConvP transfer_in;
  UpP up[2];
};
// top-level keys sorted: down, em_in, enc, mlp_input, mlp_output, mlp_par,
//                        skip, summit_mlp, summit_tb, tb_in, tm_in, transfer_in, up
static ParamsH rdAll(Cur& c) {
  ParamsH P;
  P.down[0] = rdDown(c, 64, 128);
  P.down[1] = rdDown(c, 128, 256);
  P.em_in = rdCm(c, 64, 64);
  P.enc_b1 = nf(c); P.enc_b2 = nf(c); P.enc_w1 = nf(c); P.enc_w2 = nf(c);
  P.mlp_input  = rdM1(c, 3, 64);
  P.mlp_output = rdM2(c, 64, 64, 3);
  P.mlp_par    = rdM1(c, 3, 64);
  P.skip_b = nf(c); P.skip_w = nf(c);
  P.summit_mlp = rdM1(c, 256, 256);
  P.summit_tb  = rdTb(c, 256, 256);
  P.tb_in = rdTb(c, 64, 64);
  P.tm_in = rdCm(c, 64, 64);
  P.transfer_in = rdConv(c, 64, 64, 64, false);
  P.up[0] = rdUp(c, 128, 64);
  P.up[1] = rdUp(c, 256, 128);
  return P;
}

// ---------------------------------------------------------------------------
// Launch helpers
// ---------------------------------------------------------------------------
static void lin_launch(hipStream_t st, WS& ws, const float* X, const float* Wt,
                       const float* bias, float* Y, int M, int N, int K, int act) {
  bf16* wb = (bf16*)ws_alloc(ws, sizeof(bf16) * (size_t)K * N);
  int nw = K * N;
  ku_f32_to_bf16<<<dim3((unsigned)((nw + 255) / 256)), dim3(256), 0, st>>>(Wt, wb, nw);
  dim3 g((unsigned)((M + GT_M - 1) / GT_M), (unsigned)((N + GT_N - 1) / GT_N));
  ku_gemm_wmma<<<g, dim3(128), 0, st>>>(X, wb, bias, Y, M, N, K, act);
}

static void bn_launch(hipStream_t st, WS& ws, float* X, int M, int C,
                      const float* g, const float* beta, int act) {
  float* mean = wsF(ws, C);
  float* rstd = wsF(ws, C);
  ku_bn_stats<<<dim3((unsigned)C), dim3(256), 0, st>>>(X, M, C, mean, rstd);
  int n = M * C;
  ku_bn_act<<<dim3((unsigned)((n + 255) / 256)), dim3(256), 0, st>>>(X, M, C, mean, rstd, g, beta, act);
}

static float* mlp_apply(hipStream_t st, WS& ws, const MlpP& m, const float* X, int M,
                        int act, bool norm, bool plain_last) {
  const float* cur = X;
  float* out = nullptr;
  for (int i = 0; i < m.n; i++) {
    const LinP& L = m.L[i];
    out = wsF(ws, (size_t)M * L.fout);
    bool last = (i == m.n - 1);
    if (plain_last && last) {
      lin_launch(st, ws, cur, L.w, L.b, out, M, L.fout, L.fin, 0);
    } else if (norm) {
      lin_launch(st, ws, cur, L.w, L.b, out, M, L.fout, L.fin, 0);
      bn_launch(st, ws, out, M, L.fout, L.g, L.beta, act);
    } else {
      lin_launch(st, ws, cur, L.w, L.b, out, M, L.fout, L.fin, act);
    }
    cur = out;
  }
  return out;
}

static int* knn_launch(hipStream_t st, WS& ws, const float* src, const float* dst,
                       int Ns, int Nd, int kq, int kout, int excl, int app) {
  int* idx = (int*)ws_alloc(ws, sizeof(int) * (size_t)BB * Nd * kout);
  int tot = BB * Nd;
  ku_knn<<<dim3((unsigned)((tot + 127) / 128)), dim3(128), 0, st>>>(src, dst, Ns, Nd, idx, kq, kout, excl, app);
  return idx;
}

static float* ptconv_launch(hipStream_t st, WS& ws, const ConvP& p,
                            const float* xs, const float* xd,
                            const float* ps, const float* pd,
                            const int* nbr, int Ns, int Nd, int k) {
  int co = p.co;
  float* a_src = wsF(ws, (size_t)BB * Ns * co);
  lin_launch(st, ws, xs, p.w_src, nullptr, a_src, BB * Ns, co, p.cs, 0);
  float* a_dst = wsF(ws, (size_t)BB * Nd * co);
  lin_launch(st, ws, xd, p.w_dst, nullptr, a_dst, BB * Nd, co, p.cd, 0);
  float* v = wsF(ws, (size_t)BB * Ns * co);
  lin_launch(st, ws, xs, p.w_lin, nullptr, v, BB * Ns, co, p.cs, 0);

  float* rel = wsF(ws, (size_t)BB * Nd * k * 3);
  int tot = BB * Nd * k;
  ku_rel_pos<<<dim3((unsigned)((tot + 255) / 256)), dim3(256), 0, st>>>(ps, pd, nbr, Ns, Nd, k, rel);
  float* delta = mlp_apply(st, ws, p.pos, rel, BB * Nd * k, 2, false, false);

  float* gin = wsF(ws, (size_t)BB * Nd * k * co);
  int tot2 = BB * Nd * k * co;
  ku_gamma_in<<<dim3((unsigned)((tot2 + 255) / 256)), dim3(256), 0, st>>>(a_dst, a_src, nbr, delta, Ns, Nd, k, co, gin);
  float* gamma = mlp_apply(st, ws, p.attn, gin, BB * Nd * k, 2, false, false);

  float* out = wsF(ws, (size_t)BB * Nd * co);
  ku_attn_agg<<<dim3((unsigned)(BB * Nd)), dim3((unsigned)co), 0, st>>>(gamma, v, delta, nbr, Ns, Nd, k, co, out);
  return out;
}

static float* tblock_launch(hipStream_t st, WS& ws, const TbP& t, const float* xin,
                            const float* pos, const int* nbr, int Npts) {
  float* h1 = mlp_apply(st, ws, t.mlp_in, xin, BB * Npts, 2, true, false);
  float* h2 = ptconv_launch(st, ws, t.conv, h1, h1, pos, pos, nbr, Npts, Npts, KNN_K + 1);
  return mlp_apply(st, ws, t.mlp_out, h2, BB * Npts, 0, true, false);
}

static float* cmerge_launch(hipStream_t st, WS& ws, const CmP& cm, const float* xin,
                            const float* ctxv, int Npts, int C) {
  float* cv = mlp_apply(st, ws, cm.ctx, ctxv, BB, 2, true, false);        // [B,C]
  float* xv = mlp_apply(st, ws, cm.mlp, xin, BB * Npts, 2, true, false);  // [B,Npts,C]
  int n = BB * Npts * C;
  ku_add_bcast<<<dim3((unsigned)((n + 255) / 256)), dim3(256), 0, st>>>(xv, cv, Npts, C);
  return xv;
}

// ===========================================================================
// Full UNet forward
// ===========================================================================
extern "C" void kernel_launch(void* const* d_in, const int* in_sizes, int n_in,
                              void* d_out, int out_size, void* d_ws, size_t ws_size,
                              hipStream_t stream) {
  const float* x   = (const float*)d_in[0];   // [B,N,3]
  const float* t   = (const float*)d_in[1];   // [B]
  const float* par = (const float*)d_in[2];   // [B,NPAR,3]
  Cur cur{d_in, 3, n_in};
  ParamsH P = rdAll(cur);
  WS ws{(char*)d_ws, ws_size, 0};
  hipStream_t st = stream;

  // TDM async-tensor staging demo (gfx1250 path)
  {
    float* scratch = wsF(ws, 64 * 64);
    ku_tdm_stage<<<dim3(1), dim3(64), 0, st>>>(x, scratch, 64, 64, 64);
  }

  // time embedding
  float* tctx = wsF(ws, (size_t)BB * DCTX);
  ku_time_embed<<<dim3((unsigned)((BB * DCTX + 255) / 256)), dim3(256), 0, st>>>(t, tctx, BB, DCTX);

  // PointNet context encoder: relu(par@w1+b1)@w2+b2, max-pool over points
  float* he1 = wsF(ws, (size_t)BB * NPARC * 128);
  lin_launch(st, ws, par, P.enc_w1, P.enc_b1, he1, BB * NPARC, 128, 3, 1);
  float* he2 = wsF(ws, (size_t)BB * NPARC * DCTX);
  lin_launch(st, ws, he1, P.enc_w2, P.enc_b2, he2, BB * NPARC, DCTX, 128, 0);
  float* ctx = wsF(ws, (size_t)BB * DCTX);
  ku_maxpool<<<dim3((unsigned)((BB * DCTX + 255) / 256)), dim3(256), 0, st>>>(he2, NPARC, DCTX, ctx);

  // input head: skip linear + mlp_input
  float* h = wsF(ws, (size_t)BB * NPTS0 * 64);
  lin_launch(st, ws, x, P.skip_w, P.skip_b, h, BB * NPTS0, 64, 3, 0);
  float* hi = mlp_apply(st, ws, P.mlp_input, x, BB * NPTS0, 1, true, false);
  ku_add<<<dim3((unsigned)((BB * NPTS0 * 64 + 255) / 256)), dim3(256), 0, st>>>(h, hi, BB * NPTS0 * 64);

  float* h_par = mlp_apply(st, ws, P.mlp_par, par, BB * NPARC, 1, true, false);

  // transfer_in: PTConv from par features onto x points
  int* nbr_t = knn_launch(st, ws, par, x, NPARC, NPTS0, KNN_K, KNN_K, 0, 0);
  h = ptconv_launch(st, ws, P.transfer_in, h_par, h, par, x, nbr_t, NPARC, NPTS0, KNN_K);
  h = cmerge_launch(st, ws, P.tm_in, h, tctx, NPTS0, 64);
  h = cmerge_launch(st, ws, P.em_in, h, ctx, NPTS0, 64);

  int npts[3] = { NPTS0, NPTS0 / 4, NPTS0 / 16 };
  int chan[3] = { 64, 128, 256 };
  const float* pos_l[3]; float* x_l[3]; int* self_l[3];

  self_l[0] = knn_launch(st, ws, x, x, NPTS0, NPTS0, KNN_K, KNN_K + 1, 1, 1);
  h = tblock_launch(st, ws, P.tb_in, h, x, self_l[0], NPTS0);
  pos_l[0] = x; x_l[0] = h;

  const float* pos = x;
  for (int i = 0; i < 2; i++) {
    int n0 = npts[i], n1 = npts[i + 1], c1 = chan[i + 1];
    int* fidx = (int*)ws_alloc(ws, sizeof(int) * (size_t)BB * n1);
    ku_fps<<<dim3((unsigned)BB), dim3(256), 0, st>>>(pos, n0, n1, fidx);
    float* subpos = wsF(ws, (size_t)BB * n1 * 3);
    ku_gather_rows<<<dim3((unsigned)((BB * n1 * 3 + 255) / 256)), dim3(256), 0, st>>>(pos, fidx, n0, n1, 3, subpos);
    int* nbrs = knn_launch(st, ws, pos, subpos, n0, n1, KNN_K, KNN_K, 0, 0);
    float* hf = mlp_apply(st, ws, P.down[i].mlp, h, BB * n0, 2, true, false);
    float* hm = wsF(ws, (size_t)BB * n1 * c1);
    ku_gather_mean<<<dim3((unsigned)((BB * n1 * c1 + 255) / 256)), dim3(256), 0, st>>>(hf, nbrs, n0, n1, KNN_K, c1, hm);
    pos = subpos;
    self_l[i + 1] = knn_launch(st, ws, pos, pos, n1, n1, KNN_K, KNN_K + 1, 1, 1);
    h = tblock_launch(st, ws, P.down[i].tb, hm, pos, self_l[i + 1], n1);
    h = cmerge_launch(st, ws, P.down[i].tm, h, tctx, n1, c1);
    h = cmerge_launch(st, ws, P.down[i].em, h, ctx, n1, c1);
    pos_l[i + 1] = pos; x_l[i + 1] = h;
  }

  // summit branch
  float* hs = mlp_apply(st, ws, P.summit_mlp, h, BB * npts[2], 1, false, false);
  h = tblock_launch(st, ws, P.summit_tb, hs, pos_l[2], self_l[2], npts[2]);

  // up path (each iteration rebuilds from stored skip tensors, as in source)
  for (int i = 0; i < 2; i++) {
    int dn = 2 - i, up = 1 - i;
    const UpP& U = P.up[up];
    int Ndn = npts[dn], Nup = npts[up], Cup = chan[up];
    int* nbr8 = knn_launch(st, ws, pos_l[dn], pos_l[up], Ndn, Nup, K_UPC, K_UPC, 0, 0);
    float* h_int = ptconv_launch(st, ws, U.ft, x_l[dn], x_l[up], pos_l[dn], pos_l[up],
                                 nbr8, Ndn, Nup, K_UPC);
    float* hu = mlp_apply(st, ws, U.mlp, x_l[up], BB * Nup, 2, true, false);
    int n = BB * Nup * Cup;
    ku_add<<<dim3((unsigned)((n + 255) / 256)), dim3(256), 0, st>>>(hu, h_int, n);
    h = tblock_launch(st, ws, U.tb, hu, pos_l[up], self_l[up], Nup);
    h = cmerge_launch(st, ws, U.tm, h, tctx, Nup, Cup);
    h = cmerge_launch(st, ws, U.em, h, ctx, Nup, Cup);
  }

  // output head: [64->64] BN+relu, then plain [64->3] straight into d_out
  {
    const LinP& L0 = P.mlp_output.L[0];
    const LinP& L1 = P.mlp_output.L[1];
    float* t0 = wsF(ws, (size_t)BB * NPTS0 * 64);
    lin_launch(st, ws, h, L0.w, L0.b, t0, BB * NPTS0, 64, 64, 0);
    bn_launch(st, ws, t0, BB * NPTS0, 64, L0.g, L0.beta, 1);
    lin_launch(st, ws, t0, L1.w, L1.b, (float*)d_out, BB * NPTS0, 3, 64, 0);
  }

  (void)in_sizes; (void)out_size;
}
